// HierarchicalDocumentSentiment_15006615732896
// MI455X (gfx1250) — matrix operand order repaired and checked by reading, not compiled, not run on hardware
//
#include <hip/hip_runtime.h>
#include <hip/hip_bf16.h>

// ---------------------------------------------------------------------------
// CDNA5 (gfx1250) implementation of a hierarchical BiLSTM sentiment model.
// GEMMs on v_wmma_f32_16x16x32_bf16 (wave32 WMMA, f32 accumulate); the LSTM
// recurrence double-buffers the next timestep's gathered input tile into LDS
// with GLOBAL_LOAD_ASYNC_TO_LDS_B128 (ASYNCcnt / s_wait_asynccnt).
// ---------------------------------------------------------------------------

typedef __bf16 bf16;
typedef __attribute__((ext_vector_type(16))) __bf16 v16bf;
typedef __attribute__((ext_vector_type(8)))  __bf16 v8bf;
typedef __attribute__((ext_vector_type(8)))  float  v8f;

__device__ __forceinline__ v8f wmma_bf16(v16bf a, v16bf b, v8f c) {
#if defined(__HIP_DEVICE_COMPILE__)
  // (neg_a, A, neg_b, B, c_mod, C, reuse_a, reuse_b)
  return __builtin_amdgcn_wmma_f32_16x16x32_bf16(false, a, false, b, (short)0, c, false, false);
#else
  (void)a; (void)b;
  return c;
#endif
}

__device__ __forceinline__ float sigmoidf_(float x) { return 1.f / (1.f + __expf(-x)); }

__device__ __forceinline__ v16bf pack2(v8bf lo, v8bf hi) {
  v16bf r;
#pragma unroll
  for (int i = 0; i < 8; i++) { r[i] = lo[i]; r[i + 8] = hi[i]; }
  return r;
}

__device__ __forceinline__ void wait_async0() {
#if defined(__HIP_DEVICE_COMPILE__)
  asm volatile("s_wait_asynccnt 0x0" ::: "memory");
#endif
}

// Async-DMA one timestep's input tile (16 rows x E bf16) into LDS.
// Issued by one wave; each lane moves 16-byte chunks via
// GLOBAL_LOAD_ASYNC_TO_LDS_B128 (VDST = LDS byte address, VADDR = 64b global).
template <int E>
__device__ __forceinline__ void stage_x_async(
    const bf16* __restrict__ xsrc, const int* __restrict__ ids,
    int seqbase, int T, int t, unsigned lds_base, int lane) {
  constexpr int CPR  = E / 8;              // 16B chunks per row
  constexpr int NISS = (16 * CPR) / 32;    // issues per lane
  constexpr unsigned ROWB = (unsigned)(E + 8) * 2u;  // padded LDS row stride
#pragma unroll
  for (int i = 0; i < NISS; ++i) {
    const int chunk = i * 32 + lane;
    const int row = chunk / CPR;
    const int cj  = chunk % CPR;
    const int rf  = (seqbase + row) * T + t;
    const long rowid = ids ? (long)ids[rf] : (long)rf;
    const unsigned lds_a = lds_base + (unsigned)row * ROWB + (unsigned)cj * 16u;
    const unsigned long long ga = (unsigned long long)(uintptr_t)xsrc +
                                  (unsigned long long)rowid * (unsigned)(E * 2) +
                                  (unsigned)cj * 16u;
#if defined(__HIP_DEVICE_COMPILE__)
    asm volatile("global_load_async_to_lds_b128 %0, %1, off"
                 :: "v"(lds_a), "v"(ga)
                 : "memory");
#else
    (void)lds_a; (void)ga;
#endif
  }
}

// f32 -> bf16 conversion (weights, embedding table)
__global__ __launch_bounds__(256) void cvt_kernel(const float* __restrict__ in,
                                                  bf16* __restrict__ out, int n) {
  int i = blockIdx.x * 256 + threadIdx.x;
  if (i < n) out[i] = (bf16)in[i];
}

// ---------------------------------------------------------------------------
// Fused BiLSTM direction kernel.
//   One block = 16 sequences. Per step computes g = h@Whh^T + x_t@Wih^T via
//   WMMA (8 waves x 4 N-tiles of 16 = 512 gate cols), then gates elementwise.
//   x_t tiles are async-staged (double buffered) one step ahead by wave 0.
// A layout (16x32 bf16): lane<16 -> M=lane, K=0..15;
//                        lane>=16 same rows, K=16..31.
// B layout (32x16 bf16): lane<16 -> N=lane, K=0..15; lane>=16 -> K=16..31.
//   B[k][n] = W[n][k]  ->  one contiguous 32-byte load per lane per tile.
// C/D layout: lane holds column n=lane&15 (+16*ntile), rows (lane>=16?8:0)+r.
// ---------------------------------------------------------------------------
template <int E>
__global__ __launch_bounds__(256) void lstm_dir_kernel(
    const bf16* __restrict__ Whh,   // (512,128) bf16 row-major
    const bf16* __restrict__ Wih,   // (512,E)   bf16 row-major
    const float* __restrict__ bih, const float* __restrict__ bhh,
    const bf16* __restrict__ xsrc,  // emb_bf16 (gather) or dense rows of E
    const int*  __restrict__ ids,   // token ids or nullptr
    float* __restrict__ out,        // rows of 256, write cols [outcol, outcol+128)
    int T, int reverse, int outcol) {
  constexpr int XS = E + 8;                    // padded LDS row stride (elems)
  __shared__ __align__(16) bf16 h_s[16][136];  // hidden state, bank-spread pad
  __shared__ __align__(16) bf16 x_s[2][16][XS];  // double-buffered input tile
  __shared__ float g_s[16][516];               // gate pre-activations
  __shared__ float bsum[512];

  const int tid = threadIdx.x;
  const int lane = tid & 31;
  const int wv = tid >> 5;
  const int seqbase = blockIdx.x * 16;

  for (int i = tid; i < 16 * 136; i += 256) (&h_s[0][0])[i] = (bf16)0.f;
  for (int i = tid; i < 512; i += 256) bsum[i] = bih[i] + bhh[i];
  float c_reg[8];
#pragma unroll
  for (int i = 0; i < 8; i++) c_reg[i] = 0.f;

  const unsigned xs_base = (unsigned)(uintptr_t)&x_s[0][0][0];
  constexpr unsigned XBUFB = (unsigned)(16 * XS * 2);  // bytes per buffer

  // prologue: async-stage step 0's input into buffer 0
  if (wv == 0) {
    stage_x_async<E>(xsrc, ids, seqbase, T, reverse ? (T - 1) : 0, xs_base, lane);
    wait_async0();
  }
  __syncthreads();

  const int mA   = lane & 15;
  const int kbA  = (lane >> 4) * 16;
  const int nB   = lane & 15;
  const int kbB  = (lane >> 4) * 16;
  const int mrow = (lane >> 4) * 8;

  for (int step = 0; step < T; ++step) {
    const int t = reverse ? (T - 1 - step) : step;
    const int cur = step & 1;

    // kick off next step's gather into the other buffer (overlaps WMMA below)
    if (wv == 0 && step + 1 < T) {
      const int tn = reverse ? (T - 2 - step) : (step + 1);
      stage_x_async<E>(xsrc, ids, seqbase, T, tn, xs_base + (cur ^ 1) * XBUFB, lane);
    }

    v8f acc[4] = {};

    // ---- recurrent term: h @ Whh^T (K = 128) ----
#pragma unroll
    for (int kt = 0; kt < 4; ++kt) {
      const bf16* ap = &h_s[mA][kt * 32 + kbA];
      v16bf a = pack2(*(const v8bf*)ap, *(const v8bf*)(ap + 8));
#pragma unroll
      for (int q = 0; q < 4; q++) {
        const int nt = wv + q * 8;
        v16bf b = *(const v16bf*)(Whh + (size_t)(nt * 16 + nB) * 128 + kt * 32 + kbB);
        acc[q] = wmma_bf16(a, b, acc[q]);
      }
    }

    // ---- input term: x_t @ Wih^T (K = E), x_t pre-staged in LDS ----
#pragma unroll
    for (int kt = 0; kt < E / 32; ++kt) {
      const bf16* ap = &x_s[cur][mA][kt * 32 + kbA];
      v16bf a = pack2(*(const v8bf*)ap, *(const v8bf*)(ap + 8));
#pragma unroll
      for (int q = 0; q < 4; q++) {
        const int nt = wv + q * 8;
        v16bf b = *(const v16bf*)(Wih + (size_t)(nt * 16 + nB) * E + kt * 32 + kbB);
        acc[q] = wmma_bf16(a, b, acc[q]);
      }
    }

    // scatter accumulators to LDS (column-per-lane layout -> row-major)
#pragma unroll
    for (int q = 0; q < 4; q++) {
      const int n = (wv + q * 8) * 16 + nB;
#pragma unroll
      for (int rr = 0; rr < 8; rr++) g_s[mrow + rr][n] = acc[q][rr];
    }
    __syncthreads();

    // ---- gate nonlinearity; c lives in registers across steps ----
#pragma unroll
    for (int i = 0; i < 8; i++) {
      const int idx = tid + 256 * i;     // 16 rows x 128 hidden
      const int m = idx >> 7, j = idx & 127;
      const float gi = g_s[m][j]       + bsum[j];
      const float gf = g_s[m][j + 128] + bsum[j + 128];
      const float gg = g_s[m][j + 256] + bsum[j + 256];
      const float go = g_s[m][j + 384] + bsum[j + 384];
      const float c = sigmoidf_(gf) * c_reg[i] + sigmoidf_(gi) * tanhf(gg);
      const float h = sigmoidf_(go) * tanhf(c);
      c_reg[i] = c;
      h_s[m][j] = (bf16)h;
      out[((size_t)(seqbase + m) * T + t) * 256 + outcol + j] = h;
    }
    // ensure next step's async tile has landed before anyone reads it
    if (wv == 0 && step + 1 < T) wait_async0();
    __syncthreads();
  }
}

// ---------------------------------------------------------------------------
// Attention pooling: s[t] = tanh(h[t]@W1^T + b1)@W2^T + b2; masked softmax;
// out = sum_t w[t] * h[t].  One block per sequence; scores via WMMA + LDS
// float atomics (ds_add_f32); A converted f32->bf16 at load.
// ---------------------------------------------------------------------------
__global__ __launch_bounds__(256) void attn_pool_kernel(
    const float* __restrict__ hsrc,       // (nblocks, T, 256) f32
    const bf16*  __restrict__ W1b,        // (128,256) bf16
    const float* __restrict__ b1, const float* __restrict__ W2v,
    const float* __restrict__ b2,
    const int* __restrict__ ids,          // word mask source, or nullptr
    const unsigned char* __restrict__ maskin,  // sentence mask, or nullptr
    unsigned char* __restrict__ maskout,  // write per-seq any-mask, or nullptr
    bf16* __restrict__ out_bf, float* __restrict__ out_f, int T) {
  __shared__ float s_s[64];
  __shared__ float w_s[64];
  __shared__ int   msk[64];
  const int n = blockIdx.x, tid = threadIdx.x, lane = tid & 31, wv = tid >> 5;
  const float* hbase = hsrc + (size_t)n * T * 256;

  if (tid < 64) { s_s[tid] = b2[0]; w_s[tid] = 0.f; msk[tid] = 0; }
  __syncthreads();
  if (tid < T)
    msk[tid] = ids ? (ids[(size_t)n * T + tid] != 0)
                   : (maskin[(size_t)n * T + tid] != 0);

  const int mA = lane & 15, kbA = (lane >> 4) * 16;
  const int kbB = (lane >> 4) * 16, mrow = (lane >> 4) * 8;
  const int nh = wv * 16 + (lane & 15);   // hidden col 0..127 (8 waves x 16)
  const float bb1 = b1[nh];
  const float ww2 = W2v[nh];
  const int MT = (T + 15) >> 4;

  for (int mt = 0; mt < MT; ++mt) {
    v8f acc = {};
#pragma unroll
    for (int kt = 0; kt < 8; ++kt) {      // K = 256
      v16bf a;
      const int m = mt * 16 + mA;
      if (m < T) {
        const float* p = hbase + (size_t)m * 256 + kt * 32 + kbA;
#pragma unroll
        for (int i = 0; i < 8; i++) { a[i] = (bf16)p[i]; a[i + 8] = (bf16)p[i + 8]; }
      } else {
#pragma unroll
        for (int i = 0; i < 16; i++) a[i] = (bf16)0.f;
      }
      v16bf b = *(const v16bf*)(W1b + (size_t)nh * 256 + kt * 32 + kbB);
      acc = wmma_bf16(a, b, acc);
    }
#pragma unroll
    for (int rr = 0; rr < 8; rr++) {
      const int m = mt * 16 + mrow + rr;
      if (m < T) atomicAdd(&s_s[m], tanhf(acc[rr] + bb1) * ww2);  // ds_add_f32
    }
  }
  __syncthreads();

  if (maskout && tid == 0) {
    int any = 0;
    for (int t2 = 0; t2 < T; t2++) any |= msk[t2];
    maskout[n] = (unsigned char)any;
  }
  if (tid < T && !msk[tid]) s_s[tid] = -3.0e38f;
  __syncthreads();
  if (tid < T) {
    float mx = -3.4e38f;
    for (int t2 = 0; t2 < T; t2++) mx = fmaxf(mx, s_s[t2]);
    w_s[tid] = __expf(s_s[tid] - mx);
  }
  __syncthreads();
  float ssum = 0.f;
  for (int t2 = 0; t2 < T; t2++) ssum += w_s[t2];
  const float inv = 1.f / ssum;
  const int d = tid;                       // 256 output dims
  float o = 0.f;
  for (int t2 = 0; t2 < T; t2++) o += w_s[t2] * hbase[(size_t)t2 * 256 + d];
  o *= inv;
  if (out_bf) out_bf[(size_t)n * 256 + d] = (bf16)o;
  if (out_f)  out_f[(size_t)n * 256 + d] = o;
}

// classifier: relu(doc@W1^T + b1) @ W2^T + b2  -> (64,3) logits (tiny, VALU)
__global__ __launch_bounds__(128) void classifier_kernel(
    const float* __restrict__ doc, const float* __restrict__ W1,
    const float* __restrict__ b1, const float* __restrict__ W2,
    const float* __restrict__ b2, float* __restrict__ out) {
  __shared__ float hid[128];
  const int b = blockIdx.x, j = threadIdx.x;
  const float* d = doc + (size_t)b * 256;
  float s = b1[j];
  for (int k = 0; k < 256; k++) s += d[k] * W1[j * 256 + k];
  hid[j] = fmaxf(s, 0.f);
  __syncthreads();
  if (j < 3) {
    float o = b2[j];
    for (int k = 0; k < 128; k++) o += hid[k] * W2[j * 128 + k];
    out[b * 3 + j] = o;
  }
}

extern "C" void kernel_launch(void* const* d_in, const int* in_sizes, int n_in,
                              void* d_out, int out_size, void* d_ws, size_t ws_size,
                              hipStream_t stream) {
  (void)in_sizes; (void)n_in; (void)out_size; (void)ws_size;
  const int B = 64, S = 50, W = 30;
  const int NW = B * S;  // 3200 sentences

  const int* ids = (const int*)d_in[0];

  // ---- workspace layout (256B aligned) ----
  char* ws = (char*)d_ws;
  size_t off = 0;
  auto alloc = [&](size_t bytes) -> char* {
    size_t o = off;
    off = (o + bytes + 255) & ~(size_t)255;
    return ws + o;
  };
  bf16* emb_b  = (bf16*)alloc((size_t)50000 * 128 * 2);
  bf16* wlWihF = (bf16*)alloc((size_t)65536 * 2);
  bf16* wlWhhF = (bf16*)alloc((size_t)65536 * 2);
  bf16* wlWihB = (bf16*)alloc((size_t)65536 * 2);
  bf16* wlWhhB = (bf16*)alloc((size_t)65536 * 2);
  bf16* slWihF = (bf16*)alloc((size_t)131072 * 2);
  bf16* slWhhF = (bf16*)alloc((size_t)65536 * 2);
  bf16* slWihB = (bf16*)alloc((size_t)131072 * 2);
  bf16* slWhhB = (bf16*)alloc((size_t)65536 * 2);
  bf16* waW1b  = (bf16*)alloc((size_t)32768 * 2);
  bf16* saW1b  = (bf16*)alloc((size_t)32768 * 2);
  float* wl    = (float*)alloc((size_t)NW * W * 256 * 4);
  bf16*  sent  = (bf16*) alloc((size_t)NW * 256 * 2);
  float* sl    = (float*)alloc((size_t)B * S * 256 * 4);
  float* doc   = (float*)alloc((size_t)B * 256 * 4);
  unsigned char* smask = (unsigned char*)alloc((size_t)NW);

  auto cvt = [&](const void* src, bf16* dst, int n) {
    cvt_kernel<<<(n + 255) / 256, 256, 0, stream>>>((const float*)src, dst, n);
  };
  cvt(d_in[1],  emb_b,  50000 * 128);
  cvt(d_in[2],  wlWihF, 65536);   cvt(d_in[3],  wlWhhF, 65536);
  cvt(d_in[6],  wlWihB, 65536);   cvt(d_in[7],  wlWhhB, 65536);
  cvt(d_in[10], slWihF, 131072);  cvt(d_in[11], slWhhF, 65536);
  cvt(d_in[14], slWihB, 131072);  cvt(d_in[15], slWhhB, 65536);
  cvt(d_in[18], waW1b,  32768);   cvt(d_in[22], saW1b,  32768);

  // ---- word-level BiLSTM (embedding gather fused, async-staged) ----
  lstm_dir_kernel<128><<<NW / 16, 256, 0, stream>>>(
      wlWhhF, wlWihF, (const float*)d_in[4], (const float*)d_in[5],
      emb_b, ids, wl, W, 0, 0);
  lstm_dir_kernel<128><<<NW / 16, 256, 0, stream>>>(
      wlWhhB, wlWihB, (const float*)d_in[8], (const float*)d_in[9],
      emb_b, ids, wl, W, 1, 128);

  // ---- word attention pool -> sent (bf16) + sentence mask ----
  attn_pool_kernel<<<NW, 256, 0, stream>>>(
      wl, waW1b, (const float*)d_in[19], (const float*)d_in[20],
      (const float*)d_in[21], ids, nullptr, smask, sent, nullptr, W);

  // ---- sentence-level BiLSTM ----
  lstm_dir_kernel<256><<<B / 16, 256, 0, stream>>>(
      slWhhF, slWihF, (const float*)d_in[12], (const float*)d_in[13],
      sent, nullptr, sl, S, 0, 0);
  lstm_dir_kernel<256><<<B / 16, 256, 0, stream>>>(
      slWhhB, slWihB, (const float*)d_in[16], (const float*)d_in[17],
      sent, nullptr, sl, S, 1, 128);

  // ---- sentence attention pool -> doc (f32) ----
  attn_pool_kernel<<<B, 256, 0, stream>>>(
      sl, saW1b, (const float*)d_in[23], (const float*)d_in[24],
      (const float*)d_in[25], nullptr, smask, nullptr, nullptr, doc, S);

  // ---- classifier -> logits ----
  classifier_kernel<<<B, 128, 0, stream>>>(
      doc, (const float*)d_in[26], (const float*)d_in[27],
      (const float*)d_in[28], (const float*)d_in[29], (float*)d_out);
}
